// MultiHeadAttention_79130477462149
// MI455X (gfx1250) — matrix operand order repaired and checked by reading, not compiled, not run on hardware
//
#include <hip/hip_runtime.h>

// MI455X / gfx1250 — wave32, WMMA 16x16x32 bf16 (f32 accumulate) + TDM staging.
// B=2, T=2048, C=1024, H=16, D=64.

typedef unsigned short u16;
typedef __attribute__((ext_vector_type(8)))  unsigned short u16x8;
typedef __attribute__((ext_vector_type(16))) unsigned short u16x16;
typedef __attribute__((ext_vector_type(16))) __bf16         v16bf;
typedef __attribute__((ext_vector_type(8)))  float          v8f;
typedef __attribute__((ext_vector_type(4)))  unsigned int   u32x4;
typedef __attribute__((ext_vector_type(8)))  int            i32x8;
typedef __attribute__((ext_vector_type(4)))  int            i32x4;

#if __has_builtin(__builtin_amdgcn_tensor_load_to_lds) && \
    __has_builtin(__builtin_amdgcn_s_wait_tensorcnt)
#define HAVE_TDM 1
#else
#define HAVE_TDM 0
#endif

__device__ __forceinline__ u16 f2bf(float f) {
  unsigned u = __builtin_bit_cast(unsigned, f);
  u += 0x7FFFu + ((u >> 16) & 1u);   // round-to-nearest-even
  return (u16)(u >> 16);
}

__device__ __forceinline__ v8f wmma_bf(u16x16 a, u16x16 b, v8f c) {
  return __builtin_amdgcn_wmma_f32_16x16x32_bf16(
      false, __builtin_bit_cast(v16bf, a),
      false, __builtin_bit_cast(v16bf, b),
      (short)0, c, false, false);
}

// A fragment (16xK tile, K-step 32) from a row-major [rows, ld] bf16 matrix.
// ISA 7.12.2 (16-bit A 16x32): lanes 0-15 hold K {0..7,16..23}, lanes 16-31
// hold K {8..15,24..31} for row M = lane&15. Two 16B loads.
__device__ __forceinline__ u16x16 load_fragA(const u16* base, int ld,
                                             int row0, int k0, int lane) {
  const int half = lane >> 4, r = lane & 15;
  const u16* p = base + (size_t)(row0 + r) * ld + k0 + half * 8;
  u16x8 lo = *(const u16x8*)p;
  u16x8 hi = *(const u16x8*)(p + 16);
  u16x16 f;
#pragma unroll
  for (int i = 0; i < 8; ++i) { f[i] = lo[i]; f[8 + i] = hi[i]; }
  return f;
}

// B fragment (Kx16 tile, K-step 32) from a [N][K] ("col-major") bf16 matrix.
// Col N = lane&15; lanes 0-15 hold K {0..7,16..23}, lanes 16-31 K {8..15,24..31}.
__device__ __forceinline__ u16x16 load_fragB(const u16* base, int ld,
                                             int n0, int k0, int lane) {
  const int half = lane >> 4, c = lane & 15;
  const u16* p = base + (size_t)(n0 + c) * ld + k0 + half * 8;
  u16x8 lo = *(const u16x8*)p;
  u16x8 hi = *(const u16x8*)(p + 16);
  u16x16 f;
#pragma unroll
  for (int i = 0; i < 8; ++i) { f[i] = lo[i]; f[8 + i] = hi[i]; }
  return f;
}

#if HAVE_TDM
// Issue one TDM 2-D tile load (window == tile, so no OOB clipping needed).
// D# packing per ISA cdna5 §8.3/8.4: group0 = {count=1 | lds_addr | 57b
// global_addr | type=2}; group1 = {data_size=2B, tensor dims, tile dims,
// dim0 stride}; groups 2/3 unused (2-D).
__device__ __forceinline__ void tdm_load_2d(unsigned lds_addr, const u16* gptr,
                                            unsigned dim0, unsigned dim1,
                                            unsigned tile0, unsigned tile1,
                                            unsigned long long stride0) {
  const unsigned long long ga = (unsigned long long)(uintptr_t)gptr;
  u32x4 g0;
  g0[0] = 1u;                                   // count=1, user mode
  g0[1] = lds_addr;                             // lds_addr [63:32]
  g0[2] = (unsigned)ga;                         // global_addr[31:0]
  g0[3] = (unsigned)((ga >> 32) & 0x01FFFFFFu)  // global_addr[56:32]
        | 0x80000000u;                          // type=2 ("image")
  i32x8 g1;
  g1[0] = (int)(1u << 16);                      // data_size=1 (2 bytes)
  g1[1] = (int)((dim0 & 0xFFFFu) << 16);        // tensor_dim0[15:0] @ bits63:48
  g1[2] = (int)((dim0 >> 16) | ((dim1 & 0xFFFFu) << 16));
  g1[3] = (int)((dim1 >> 16) | (tile0 << 16));  // tile_dim0 @ bits127:112
  g1[4] = (int)(tile1 & 0xFFFFu);               // tile_dim1; tile_dim2=0
  g1[5] = (int)(unsigned)stride0;               // tensor_dim0_stride[31:0]
  g1[6] = (int)((stride0 >> 32) & 0xFFFFu);     // stride0[47:32]; stride1=0
  g1[7] = 0;
  i32x4 z4 = {0, 0, 0, 0};
#if __clang_major__ >= 23
  i32x8 z8 = {0, 0, 0, 0, 0, 0, 0, 0};
  __builtin_amdgcn_tensor_load_to_lds(g0, g1, z4, z4, z8, 0);
#else
  __builtin_amdgcn_tensor_load_to_lds(g0, g1, z4, z4, 0);
#endif
}
__device__ __forceinline__ unsigned lds_byte_off(const void* p) {
  // LDS aperture maps low 32 bits of the generic address to the LDS offset.
  return (unsigned)(uintptr_t)p;
}
#endif

// ---------------- prep: fp32 -> bf16 staging ----------------

__global__ void cvt_x_kernel(const float* __restrict__ in, u16* __restrict__ out, int n) {
  int i = blockIdx.x * blockDim.x + threadIdx.x;
  if (i < n) out[i] = f2bf(in[i]);
}

// W[h][c][d] (f32, [16,1024,64]) -> Wt[n][c] bf16 with n = h*64+d  ([1024][1024])
__global__ void cvt_wqkv_kernel(const float* __restrict__ W, u16* __restrict__ o) {
  int i = blockIdx.x * blockDim.x + threadIdx.x;   // 1048576 elems
  int n = i >> 10, c = i & 1023;
  o[i] = f2bf(W[((size_t)(n >> 6) * 1024 + c) * 64 + (n & 63)]);
}

// Wp[k][n] (f32 [1024,1024]) -> WpT[n][k] bf16
__global__ void cvt_wp_kernel(const float* __restrict__ W, u16* __restrict__ o) {
  int i = blockIdx.x * blockDim.x + threadIdx.x;
  int n = i >> 10, c = i & 1023;
  o[i] = f2bf(W[(size_t)c * 1024 + n]);
}

// ---------------- QKV projection GEMM (ping-pong pipelined) ----------------
__global__ __launch_bounds__(256) void gemm_qkv_kernel(
    const u16* __restrict__ x,  const u16* __restrict__ wq,
    const u16* __restrict__ wk, const u16* __restrict__ wv,
    u16* __restrict__ q, u16* __restrict__ k, u16* __restrict__ v) {
  const int lane = threadIdx.x & 31, wave = threadIdx.x >> 5;
  const int n0 = blockIdx.x * 64;
  const int m0 = blockIdx.y * 128 + wave * 16;
  const int which = blockIdx.z;
  const u16* w = (which == 0) ? wq : (which == 1) ? wk : wv;

  v8f acc[4];
#pragma unroll
  for (int n = 0; n < 4; ++n)
#pragma unroll
    for (int i = 0; i < 8; ++i) acc[n][i] = 0.f;

  // ping-pong double buffer: no register rotation copies
  u16x16 a0 = load_fragA(x, 1024, m0, 0, lane);
  u16x16 b0[4];
#pragma unroll
  for (int n = 0; n < 4; ++n) b0[n] = load_fragB(w, 1024, n0 + n * 16, 0, lane);

#pragma unroll 1
  for (int k0 = 0; k0 < 1024; k0 += 64) {
    __builtin_prefetch((const void*)(x + (size_t)m0 * 1024 + k0 + 192), 0, 1);
    u16x16 a1 = load_fragA(x, 1024, m0, k0 + 32, lane);
    u16x16 b1[4];
#pragma unroll
    for (int n = 0; n < 4; ++n)
      b1[n] = load_fragB(w, 1024, n0 + n * 16, k0 + 32, lane);
#pragma unroll
    for (int n = 0; n < 4; ++n) acc[n] = wmma_bf(a0, b0[n], acc[n]);
    if (k0 + 64 < 1024) {
      a0 = load_fragA(x, 1024, m0, k0 + 64, lane);
#pragma unroll
      for (int n = 0; n < 4; ++n)
        b0[n] = load_fragB(w, 1024, n0 + n * 16, k0 + 64, lane);
    }
#pragma unroll
    for (int n = 0; n < 4; ++n) acc[n] = wmma_bf(a1, b1[n], acc[n]);
  }

  const int half = lane >> 4, c = lane & 15;
#pragma unroll
  for (int n = 0; n < 4; ++n) {
    const int nn = n0 + n * 16 + c;
    const int h = nn >> 6, d = nn & 63;
#pragma unroll
    for (int r = 0; r < 8; ++r) {
      const int m = m0 + r + 8 * half;
      const int b = m >> 11, t = m & 2047;      // T = 2048
      const int g = h * 2 + b;                  // (h,b) flat, B = 2
      const u16 val = f2bf(acc[n][r]);
      if (which == 0)      q[((size_t)g * 2048 + t) * 64 + d] = val;
      else if (which == 1) k[((size_t)g * 2048 + t) * 64 + d] = val;
      else                 v[((size_t)g * 64 + d) * 2048 + t] = val;   // transposed
    }
  }
}

// ---------------- fused flash attention with TDM-staged K/V ----------------
// grid = (T/64, H*B), block = 128 (4 waves). Each wave: 16 query rows.
// q,k: [HB, T, 64] bf16; v: [HB, 64, T] bf16; o: flat [H,B,T,D] bf16.
__global__ __launch_bounds__(128) void attn_kernel(
    const u16* __restrict__ q, const u16* __restrict__ k,
    const u16* __restrict__ v, u16* __restrict__ o) {
  __shared__ u16 kbuf[2][32 * 64];   // double-buffered K tile (s-major, ld=64)
  __shared__ u16 vbuf[2][64 * 32];   // double-buffered V tile (d-major, ld=32)
  __shared__ u16 pbuf[4][16 * 40];   // per-wave P tile, padded ld=40

  const int lane = threadIdx.x & 31, wave = threadIdx.x >> 5;
  const int hb = blockIdx.y;
  const int row0 = blockIdx.x * 64 + wave * 16;
  const u16* qb = q + (size_t)hb * 2048 * 64;
  const u16* kb = k + (size_t)hb * 2048 * 64;
  const u16* vb = v + (size_t)hb * 64 * 2048;
  const int half = lane >> 4, c = lane & 15;

  // Q tile (16x64) held in registers as two A fragments (K=0..31, 32..63)
  const u16x16 aq0 = load_fragA(qb, 64, row0, 0,  lane);
  const u16x16 aq1 = load_fragA(qb, 64, row0, 32, lane);

  float mrow[8], lrow[8];
  v8f O[4];
#pragma unroll
  for (int r = 0; r < 8; ++r) { mrow[r] = -3.0e38f; lrow[r] = 0.f; }
#pragma unroll
  for (int n = 0; n < 4; ++n)
#pragma unroll
    for (int i = 0; i < 8; ++i) O[n][i] = 0.f;

  const float scale = 0.125f;   // 1/sqrt(64)
  const int NB = 64;            // 2048 / 32 key blocks

#if HAVE_TDM
  if (wave == 0) {  // TDM issue is per-wave; one wave drives the DMA
    tdm_load_2d(lds_byte_off(&kbuf[0][0]), kb,  64, 32, 64, 32, 64);
    tdm_load_2d(lds_byte_off(&vbuf[0][0]), vb,  32, 64, 32, 64, 2048);
  }
#endif

#pragma unroll 1
  for (int ib = 0; ib < NB; ++ib) {
    const int buf = ib & 1;
    const int s0 = ib * 32;
    __syncthreads();   // (a) everyone done with buffers from block ib-1
#if HAVE_TDM
    if (wave == 0) {
      if (ib + 1 < NB) {
        const u16* kn = kb + (size_t)(s0 + 32) * 64;
        const u16* vn = vb + (s0 + 32);
        tdm_load_2d(lds_byte_off(&kbuf[buf ^ 1][0]), kn, 64, 32, 64, 32, 64);
        tdm_load_2d(lds_byte_off(&vbuf[buf ^ 1][0]), vn, 32, 64, 32, 64, 2048);
        __builtin_amdgcn_s_wait_tensorcnt(2);   // block ib's two tiles landed
      } else {
        __builtin_amdgcn_s_wait_tensorcnt(0);
      }
    }
#else
    // fallback staging: cooperative copy, 2048 u16 per tile
    for (int i = threadIdx.x; i < 2048; i += 128) {
      kbuf[buf][i] = kb[(size_t)(s0 + (i >> 6)) * 64 + (i & 63)];
      vbuf[buf][i] = vb[(size_t)(i >> 5) * 2048 + s0 + (i & 31)];
    }
#endif
    __syncthreads();   // (b) K/V tiles for block ib visible to all waves

    // S = Q * K^T for 16x32 keys, fragments out of LDS (ds_load_b128)
    v8f S0, S1;
#pragma unroll
    for (int i = 0; i < 8; ++i) { S0[i] = 0.f; S1[i] = 0.f; }
    u16x16 bk;
    bk = load_fragB(&kbuf[buf][0], 64, 0,  0,  lane); S0 = wmma_bf(aq0, bk, S0);
    bk = load_fragB(&kbuf[buf][0], 64, 0,  32, lane); S0 = wmma_bf(aq1, bk, S0);
    bk = load_fragB(&kbuf[buf][0], 64, 16, 0,  lane); S1 = wmma_bf(aq0, bk, S1);
    bk = load_fragB(&kbuf[buf][0], 64, 16, 32, lane); S1 = wmma_bf(aq1, bk, S1);

    // online softmax: row M = r + 8*half lives across 16 lanes of a half
    float alpha[8];
#pragma unroll
    for (int r = 0; r < 8; ++r) {
      float a0 = S0[r] * scale, a1 = S1[r] * scale;
      float mx = fmaxf(a0, a1);
#pragma unroll
      for (int msk = 1; msk < 16; msk <<= 1) mx = fmaxf(mx, __shfl_xor(mx, msk, 32));
      const float nm = fmaxf(mrow[r], mx);
      const float p0 = __expf(a0 - nm), p1 = __expf(a1 - nm);
      float ps = p0 + p1;
#pragma unroll
      for (int msk = 1; msk < 16; msk <<= 1) ps += __shfl_xor(ps, msk, 32);
      alpha[r] = __expf(mrow[r] - nm);
      lrow[r] = lrow[r] * alpha[r] + ps;
      mrow[r] = nm;
      pbuf[wave][(r + 8 * half) * 40 + c]      = f2bf(p0);
      pbuf[wave][(r + 8 * half) * 40 + 16 + c] = f2bf(p1);
    }
#pragma unroll
    for (int n = 0; n < 4; ++n)
#pragma unroll
      for (int r = 0; r < 8; ++r) O[n][r] *= alpha[r];

    __syncthreads();   // (c) P tile staged
    const u16x16 ap = load_fragA(&pbuf[wave][0], 40, 0, 0, lane);
#pragma unroll
    for (int n = 0; n < 4; ++n) {
      u16x16 bv = load_fragB(&vbuf[buf][0], 32, n * 16, 0, lane);
      O[n] = wmma_bf(ap, bv, O[n]);
    }
  }

  // epilogue: normalize, write flat [H,B,T,D] (raw-reshape-faithful layout)
  const size_t obase = ((size_t)hb * 2048 + row0) * 64;
#pragma unroll
  for (int n = 0; n < 4; ++n)
#pragma unroll
    for (int r = 0; r < 8; ++r) {
      const float val = O[n][r] / lrow[r];
      o[obase + (size_t)(r + 8 * half) * 64 + n * 16 + c] = f2bf(val);
    }
}

// ---------------- output projection GEMM (+bias, fp32 out) ----------------
__global__ __launch_bounds__(256) void gemm_proj_kernel(
    const u16* __restrict__ a, const u16* __restrict__ w,
    const float* __restrict__ bias, float* __restrict__ out) {
  const int lane = threadIdx.x & 31, wave = threadIdx.x >> 5;
  const int n0 = blockIdx.x * 64;
  const int m0 = blockIdx.y * 128 + wave * 16;

  v8f acc[4];
#pragma unroll
  for (int n = 0; n < 4; ++n)
#pragma unroll
    for (int i = 0; i < 8; ++i) acc[n][i] = 0.f;

  u16x16 a0 = load_fragA(a, 1024, m0, 0, lane);
  u16x16 b0[4];
#pragma unroll
  for (int n = 0; n < 4; ++n) b0[n] = load_fragB(w, 1024, n0 + n * 16, 0, lane);

#pragma unroll 1
  for (int k0 = 0; k0 < 1024; k0 += 64) {
    __builtin_prefetch((const void*)(a + (size_t)m0 * 1024 + k0 + 192), 0, 1);
    u16x16 a1 = load_fragA(a, 1024, m0, k0 + 32, lane);
    u16x16 b1[4];
#pragma unroll
    for (int n = 0; n < 4; ++n)
      b1[n] = load_fragB(w, 1024, n0 + n * 16, k0 + 32, lane);
#pragma unroll
    for (int n = 0; n < 4; ++n) acc[n] = wmma_bf(a0, b0[n], acc[n]);
    if (k0 + 64 < 1024) {
      a0 = load_fragA(a, 1024, m0, k0 + 64, lane);
#pragma unroll
      for (int n = 0; n < 4; ++n)
        b0[n] = load_fragB(w, 1024, n0 + n * 16, k0 + 64, lane);
    }
#pragma unroll
    for (int n = 0; n < 4; ++n) acc[n] = wmma_bf(a1, b1[n], acc[n]);
  }

  const int half = lane >> 4, c = lane & 15;
#pragma unroll
  for (int n = 0; n < 4; ++n) {
    const int nn = n0 + n * 16 + c;
    const float bv = bias[nn];
#pragma unroll
    for (int r = 0; r < 8; ++r) {
      const int m = m0 + r + 8 * half;
      out[(size_t)m * 1024 + nn] = acc[n][r] + bv;
    }
  }
}

// ---------------- launch ----------------
extern "C" void kernel_launch(void* const* d_in, const int* in_sizes, int n_in,
                              void* d_out, int out_size, void* d_ws, size_t ws_size,
                              hipStream_t stream) {
  (void)in_sizes; (void)n_in; (void)out_size; (void)ws_size;
  const float* x  = (const float*)d_in[0];
  const float* Wq = (const float*)d_in[1];
  const float* Wk = (const float*)d_in[2];
  const float* Wv = (const float*)d_in[3];
  const float* Wp = (const float*)d_in[4];
  const float* bp = (const float*)d_in[5];
  float* out = (float*)d_out;

  // workspace carve-out (u16 elements), total 48 MiB
  u16* ws  = (u16*)d_ws;
  u16* xbf = ws;                  // [4096,1024]
  u16* q   = xbf + 4194304;       // [H,B,T,D] flat
  u16* k   = q   + 4194304;
  u16* v   = k   + 4194304;       // [H,B,D,T] flat (transposed)
  u16* ao  = v   + 4194304;       // attention out, flat [H,B,T,D] == [4096,1024]
  u16* wqT = ao  + 4194304;       // [1024(n)][1024(k)]
  u16* wkT = wqT + 1048576;
  u16* wvT = wkT + 1048576;
  u16* wpT = wvT + 1048576;

  cvt_x_kernel   <<<4194304 / 256, 256, 0, stream>>>(x, xbf, 4194304);
  cvt_wqkv_kernel<<<1048576 / 256, 256, 0, stream>>>(Wq, wqT);
  cvt_wqkv_kernel<<<1048576 / 256, 256, 0, stream>>>(Wk, wkT);
  cvt_wqkv_kernel<<<1048576 / 256, 256, 0, stream>>>(Wv, wvT);
  cvt_wp_kernel  <<<1048576 / 256, 256, 0, stream>>>(Wp, wpT);

  gemm_qkv_kernel <<<dim3(16, 32, 3), 256, 0, stream>>>(xbf, wqT, wkT, wvT, q, k, v);
  attn_kernel     <<<dim3(32, 32),    128, 0, stream>>>(q, k, v, ao);
  gemm_proj_kernel<<<dim3(16, 32),    256, 0, stream>>>(ao, wpT, bp, out);
}
